// TrajPreLocalAttnLong_82042465288739
// MI455X (gfx1250) — compile-verified, matrix operand order checked by reading, not run on hardware
//
#include <hip/hip_runtime.h>
#include <hip/hip_bf16.h>
#include <math.h>

// ---------------------------------------------------------------------------
// TrajPreLocalAttnLong on MI455X (gfx1250, wave32, WMMA)
//
//  K0 convert    : f32 weights -> bf16 workspace copies (fc_W padded to 20096 rows)
//  K1 embed      : gather loc/tim/user embeddings -> masked bf16 history/current
//  K2 gemm_xw    : gx = x @ Wih^T, register-blocked 4 M-tiles/wave (B-tile reuse)
//  K3 gru        : sequential recurrence, 4 WGs x 16 batch rows, h in LDS,
//                  A operand (h) hoisted to 16 v16bf regs per step, WMMA N-tiles
//  K4 attn       : per-batch energies/context WMMA with per-wave A-hoist,
//                  masked softmax in LDS, writes fcin = [hs | context] bf16
//  K5 fc         : y = fcin @ fc_W^T + b, register-blocked 4 M-tiles/wave
//  K6 lse        : per-row max / log-sum-exp;  K7 finalize: y - max - lse
// ---------------------------------------------------------------------------

#define B_      64
#define MAXSEQ_ 384
#define MAXH_   256
#define MAXT_   128
#define LOC_    20000
#define LE_     256
#define TE_     32
#define UE_     64
#define D_      352
#define H_      512
#define G3_     1536            // 3*H
#define NPAD_   20096           // 157 * 128, fc_W row padding for full tiles

typedef __bf16 bf16_t;
typedef __attribute__((ext_vector_type(16))) __bf16 v16bf;
typedef __attribute__((ext_vector_type(8)))  float  v8f;

// ---- WMMA helpers (CDNA5 16x16x32 bf16 layouts, wave32) -------------------
// A operand: lane holds row m = lane&15; lanes 0-15 carry K {0..7,16..23},
// lanes 16-31 carry K {8..15,24..31} (ISA 7.12.2, 16-bit A 16x32).
static __device__ __forceinline__ v16bf load_tileA(const bf16_t* p, int ld, int lane) {
  int m  = lane & 15;
  int kb = (lane >> 4) << 3;
  const bf16_t* row = p + (long long)m * ld;
  v16bf r;
#pragma unroll
  for (int i = 0; i < 8; ++i) r[i] = row[kb + i];
#pragma unroll
  for (int i = 0; i < 8; ++i) r[8 + i] = row[kb + 16 + i];
  return r;
}

// B operand when memory is K-major (element (k,n) at p[k*ld + n]).
static __device__ __forceinline__ v16bf load_tileB_kmajor(const bf16_t* p, int ld, int lane) {
  int n  = lane & 15;
  int kb = (lane >> 4) << 3;
  v16bf r;
#pragma unroll
  for (int i = 0; i < 8; ++i) r[i]     = p[(long long)(kb + i) * ld + n];
#pragma unroll
  for (int i = 0; i < 8; ++i) r[8 + i] = p[(long long)(kb + 16 + i) * ld + n];
  return r;
}

static __device__ __forceinline__ v8f wmma_bf16(v16bf a, v16bf b, v8f c) {
  return __builtin_amdgcn_wmma_f32_16x16x32_bf16(false, a, false, b, (short)0, c,
                                                 false, false);
}

// C/D layout: VGPR i -> M = i + (lane>=16 ? 8 : 0), N = lane&15.
static __device__ __forceinline__ void store_tileC(float* p, int ld, int lane, v8f c) {
  int n  = lane & 15;
  int mb = (lane >> 4) << 3;
#pragma unroll
  for (int i = 0; i < 8; ++i) p[(long long)(mb + i) * ld + n] = c[i];
}

// ---------------------------------------------------------------------------
__global__ void k_convert(const float* __restrict__ s, bf16_t* __restrict__ d,
                          long long n) {
  long long i  = (long long)blockIdx.x * blockDim.x + threadIdx.x;
  long long st = (long long)gridDim.x * blockDim.x;
  for (; i < n; i += st) d[i] = (bf16_t)s[i];
}

__global__ void k_convert_pad(const float* __restrict__ s, bf16_t* __restrict__ d,
                              int rows_src, int rows_dst, int cols) {
  long long n  = (long long)rows_dst * cols;
  long long i  = (long long)blockIdx.x * blockDim.x + threadIdx.x;
  long long st = (long long)gridDim.x * blockDim.x;
  for (; i < n; i += st) {
    int r = (int)(i / cols);
    d[i] = (r < rows_src) ? (bf16_t)s[i] : (bf16_t)0.f;
  }
}

// ---------------------------------------------------------------------------
__global__ void k_embed(const int* __restrict__ loc, const int* __restrict__ tim,
                        const int* __restrict__ uid, const int* __restrict__ seq_len,
                        const int* __restrict__ target_len,
                        const float* __restrict__ eloc, const float* __restrict__ etim,
                        const float* __restrict__ euser,
                        bf16_t* __restrict__ xh, bf16_t* __restrict__ xc) {
  const long long nh = (long long)B_ * MAXH_ * D_;
  const long long nc = (long long)B_ * MAXT_ * D_;
  long long i  = (long long)blockIdx.x * blockDim.x + threadIdx.x;
  long long st = (long long)gridDim.x * blockDim.x;
  for (; i < nh + nc; i += st) {
    int b, t, d, tabs; bool m; bf16_t* dst; long long di;
    if (i < nh) {
      b = (int)(i / (MAXH_ * D_));
      int rem = (int)(i % (MAXH_ * D_));
      t = rem / D_; d = rem % D_;
      int hlen = seq_len[b] - target_len[b];
      m = (t < hlen); tabs = t; dst = xh; di = i;
    } else {
      long long j = i - nh;
      b = (int)(j / (MAXT_ * D_));
      int rem = (int)(j % (MAXT_ * D_));
      t = rem / D_; d = rem % D_;
      int hlen = seq_len[b] - target_len[b];
      int ta = hlen + t; if (ta > MAXSEQ_ - 1) ta = MAXSEQ_ - 1;
      m = (t < target_len[b]); tabs = ta; dst = xc; di = j;
    }
    float v;
    if (d < LE_)              v = eloc[(long long)loc[b * MAXSEQ_ + tabs] * LE_ + d];
    else if (d < LE_ + TE_)   v = etim[(long long)tim[b * MAXSEQ_ + tabs] * TE_ + (d - LE_)];
    else                      v = euser[(long long)uid[b] * UE_ + (d - LE_ - TE_)];
    dst[di] = m ? (bf16_t)v : (bf16_t)0.f;
  }
}

__global__ void k_mask(const int* __restrict__ target_len, float* __restrict__ omask) {
  int i = blockIdx.x * blockDim.x + threadIdx.x;
  if (i < B_ * MAXT_) omask[i] = ((i & (MAXT_ - 1)) < target_len[i >> 7]) ? 1.f : 0.f;
}

// ---------------------------------------------------------------------------
// gx = x @ Wih^T : C[M,G3] = A[M,D] @ W[G3,D]^T. Register-blocked: each wave
// owns one N-tile and 4 M-tiles -> B tile loaded once per k-step, 4 WMMAs.
// Grid: (G3/128, M/64), block 256 (8 waves).
__global__ void k_gemm_xw(const bf16_t* __restrict__ A, const bf16_t* __restrict__ W,
                          float* __restrict__ C) {
  int lane = threadIdx.x & 31, wave = threadIdx.x >> 5;
  long long m0 = (long long)blockIdx.y * 64;
  int n0 = (blockIdx.x * 8 + wave) * 16;
  v8f acc0 = {}, acc1 = {}, acc2 = {}, acc3 = {};
  const bf16_t* Ap = A + m0 * D_;
  const bf16_t* Wp = W + (long long)n0 * D_;
  for (int k = 0; k < D_; k += 32) {
    v16bf b  = load_tileA(Wp + k, D_, lane);
    v16bf a0 = load_tileA(Ap + k, D_, lane);
    v16bf a1 = load_tileA(Ap + 16 * D_ + k, D_, lane);
    v16bf a2 = load_tileA(Ap + 32 * D_ + k, D_, lane);
    v16bf a3 = load_tileA(Ap + 48 * D_ + k, D_, lane);
    acc0 = wmma_bf16(a0, b, acc0);
    acc1 = wmma_bf16(a1, b, acc1);
    acc2 = wmma_bf16(a2, b, acc2);
    acc3 = wmma_bf16(a3, b, acc3);
  }
  store_tileC(C + (m0 +  0) * G3_ + n0, G3_, lane, acc0);
  store_tileC(C + (m0 + 16) * G3_ + n0, G3_, lane, acc1);
  store_tileC(C + (m0 + 32) * G3_ + n0, G3_, lane, acc2);
  store_tileC(C + (m0 + 48) * G3_ + n0, G3_, lane, acc3);
}

// ---------------------------------------------------------------------------
// Sequential GRU: grid = 4 WGs x 16 batch rows. h lives in LDS (f32 master +
// bf16 WMMA operand). Per step the full A operand (16x512 bf16) is hoisted
// into 16 v16bf registers and reused across all 12 N-tiles of this wave.
__global__ void k_gru(const float* __restrict__ gx, const bf16_t* __restrict__ Whh,
                      const float* __restrict__ bih, const float* __restrict__ bhh,
                      const int* __restrict__ seq_len, const int* __restrict__ target_len,
                      bf16_t* __restrict__ hid_bf, int T, int is_enc) {
  extern __shared__ char smem[];
  float*  h_f = (float*)smem;                                   // 16 x 512 f32
  float*  gh  = (float*)(smem + 16 * H_ * 4);                   // 16 x 1536 f32
  bf16_t* h_b = (bf16_t*)(smem + 16 * H_ * 4 + 16 * G3_ * 4);   // 16 x 512 bf16
  const int tid = threadIdx.x, lane = tid & 31, wave = tid >> 5;
  const int bg = blockIdx.x * 16;

  for (int i = tid; i < 16 * H_; i += 256) { h_f[i] = 0.f; h_b[i] = (bf16_t)0.f; }
  __syncthreads();

  for (int t = 0; t < T; ++t) {
    // Hoist A = h (16x512) from LDS once per step: 16 k-slices.
    v16bf ha[16];
#pragma unroll
    for (int kk = 0; kk < 16; ++kk)
      ha[kk] = load_tileA(h_b + kk * 32, H_, lane);

    // gh = h @ Whh^T : 96 N-tiles over 8 waves, 12 per wave.
    for (int nt = wave; nt < G3_ / 16; nt += 8) {
      int n0 = nt * 16;
      v8f acc = {};
      const bf16_t* Wp = Whh + (long long)n0 * H_;
#pragma unroll
      for (int kk = 0; kk < 16; ++kk)
        acc = wmma_bf16(ha[kk], load_tileA(Wp + kk * 32, H_, lane), acc);
      store_tileC(gh + n0, G3_, lane, acc);
    }
    __syncthreads();

    for (int i = tid; i < 16 * H_; i += 256) {
      int bi = i >> 9, hc = i & (H_ - 1);
      int bb = bg + bi;
      const float* gxp = gx + ((long long)bb * T + t) * G3_;
      const float* ghp = gh + bi * G3_;
      float xr = gxp[hc]          + bih[hc];
      float xz = gxp[H_ + hc]     + bih[H_ + hc];
      float xn = gxp[2 * H_ + hc] + bih[2 * H_ + hc];
      float hr = ghp[hc]          + bhh[hc];
      float hz = ghp[H_ + hc]     + bhh[H_ + hc];
      float hn = ghp[2 * H_ + hc] + bhh[2 * H_ + hc];
      float r  = 1.f / (1.f + __expf(-(xr + hr)));
      float z  = 1.f / (1.f + __expf(-(xz + hz)));
      float nn = tanhf(xn + r * hn);
      float hprev = h_f[i];
      float hnew  = (1.f - z) * nn + z * hprev;
      int lenv = is_enc ? (seq_len[bb] - target_len[bb]) : target_len[bb];
      bool m = (t < lenv);
      float hout = m ? hnew : hprev;
      h_f[i] = hout;
      h_b[i] = (bf16_t)hout;
      hid_bf[((long long)bb * T + t) * H_ + hc] = (bf16_t)(m ? hnew : 0.f);
    }
    __syncthreads();
  }
}

// ---------------------------------------------------------------------------
// Attention per batch. 8 waves = 8 M-tile rows of 16; each wave hoists its A
// operand once and sweeps all N-tiles. energies -> masked softmax -> context.
__global__ void k_attn(const bf16_t* __restrict__ hh_bf, const bf16_t* __restrict__ hs_bf,
                       const int* __restrict__ seq_len, const int* __restrict__ target_len,
                       bf16_t* __restrict__ fcin) {
  extern __shared__ char smem[];
  float*  e   = (float*)smem;                         // 128 x 256 f32
  bf16_t* abf = (bf16_t*)(smem + MAXT_ * MAXH_ * 4);  // 128 x 256 bf16
  const int b = blockIdx.x, tid = threadIdx.x, lane = tid & 31, wave = tid >> 5;
  const bf16_t* hs = hs_bf + (long long)b * MAXT_ * H_;
  const bf16_t* hh = hh_bf + (long long)b * MAXH_ * H_;

  // fcin[:, 0:512] = hidden_state (bf16)
  for (int i = tid; i < MAXT_ * H_; i += 256) {
    int t = i >> 9, h = i & (H_ - 1);
    fcin[((long long)b * MAXT_ + t) * (2 * H_) + h] = hs[i];
  }

  // energies: M=128 (t), N=256 (s), K=512. mt = wave; A hoisted (16 slices).
  {
    const int mt = wave;
    v16bf a[16];
#pragma unroll
    for (int kk = 0; kk < 16; ++kk)
      a[kk] = load_tileA(hs + mt * 16 * H_ + kk * 32, H_, lane);
    for (int nt = 0; nt < 16; ++nt) {
      v8f acc = {};
#pragma unroll
      for (int kk = 0; kk < 16; ++kk)
        acc = wmma_bf16(a[kk], load_tileA(hh + nt * 16 * H_ + kk * 32, H_, lane), acc);
      store_tileC(e + mt * 16 * MAXH_ + nt * 16, MAXH_, lane, acc);
    }
  }
  __syncthreads();

  // masked softmax over s, one thread per t-row
  int hlen = seq_len[b] - target_len[b];
  if (tid < MAXT_) {
    float* row = e + tid * MAXH_;
    float mx = -1e30f;
    for (int s = 0; s < MAXH_; ++s) {
      float v = (s < hlen) ? row[s] : -1e9f;
      row[s] = v;
      mx = fmaxf(mx, v);
    }
    float sum = 0.f;
    for (int s = 0; s < MAXH_; ++s) sum += __expf(row[s] - mx);
    float inv = 1.f / sum;
    for (int s = 0; s < MAXH_; ++s)
      abf[tid * MAXH_ + s] = (bf16_t)(__expf(row[s] - mx) * inv);
  }
  __syncthreads();

  // context: M=128 (t), N=512 (h), K=256 (s). mt = wave; A hoisted (8 slices).
  {
    const int mt = wave;
    v16bf a[8];
#pragma unroll
    for (int kk = 0; kk < 8; ++kk)
      a[kk] = load_tileA(abf + mt * 16 * MAXH_ + kk * 32, MAXH_, lane);
    for (int nt = 0; nt < 32; ++nt) {
      v8f acc = {};
#pragma unroll
      for (int kk = 0; kk < 8; ++kk)
        acc = wmma_bf16(a[kk],
                        load_tileB_kmajor(hh + (long long)(kk * 32) * H_ + nt * 16, H_, lane),
                        acc);
      int n = lane & 15, mb = (lane >> 4) << 3;
#pragma unroll
      for (int i = 0; i < 8; ++i)
        fcin[((long long)b * MAXT_ + mt * 16 + mb + i) * (2 * H_) + H_ + nt * 16 + n] =
            (bf16_t)acc[i];
    }
  }
}

// ---------------------------------------------------------------------------
// y = fcin @ fc_W^T + b : M=8192, N=20000 (padded 20096), K=1024.
// Register-blocked 4 M-tiles per wave: weight tile loaded once per k-step.
// Grid: (NPAD/128, M/64), block 256.
__global__ void k_fc(const bf16_t* __restrict__ A, const bf16_t* __restrict__ W,
                     const float* __restrict__ bias, float* __restrict__ Y) {
  int lane = threadIdx.x & 31, wave = threadIdx.x >> 5;
  long long m0 = (long long)blockIdx.y * 64;
  int n0 = (blockIdx.x * 8 + wave) * 16;
  v8f acc0 = {}, acc1 = {}, acc2 = {}, acc3 = {};
  const bf16_t* Ap = A + m0 * (2 * H_);
  const bf16_t* Wp = W + (long long)n0 * (2 * H_);
  for (int k = 0; k < 2 * H_; k += 32) {
    __builtin_prefetch(Wp + k + 512, 0, 0);   // global_prefetch_b8, L2 streaming
    v16bf bt = load_tileA(Wp + k, 2 * H_, lane);
    v16bf a0 = load_tileA(Ap + k, 2 * H_, lane);
    v16bf a1 = load_tileA(Ap + 16 * (2 * H_) + k, 2 * H_, lane);
    v16bf a2 = load_tileA(Ap + 32 * (2 * H_) + k, 2 * H_, lane);
    v16bf a3 = load_tileA(Ap + 48 * (2 * H_) + k, 2 * H_, lane);
    acc0 = wmma_bf16(a0, bt, acc0);
    acc1 = wmma_bf16(a1, bt, acc1);
    acc2 = wmma_bf16(a2, bt, acc2);
    acc3 = wmma_bf16(a3, bt, acc3);
  }
  int n = n0 + (lane & 15);
  if (n < LOC_) {
    int mb = (lane >> 4) << 3;
    float bb = bias[n];
#pragma unroll
    for (int i = 0; i < 8; ++i) {
      Y[(m0 +  0 + mb + i) * (long long)LOC_ + n] = acc0[i] + bb;
      Y[(m0 + 16 + mb + i) * (long long)LOC_ + n] = acc1[i] + bb;
      Y[(m0 + 32 + mb + i) * (long long)LOC_ + n] = acc2[i] + bb;
      Y[(m0 + 48 + mb + i) * (long long)LOC_ + n] = acc3[i] + bb;
    }
  }
}

// ---------------------------------------------------------------------------
__global__ void k_lse(const float* __restrict__ Y, float* __restrict__ stats) {
  __shared__ float sd[256];
  const long long row = blockIdx.x;
  const float* p = Y + row * LOC_;
  float mx = -1e30f;
  for (int i = threadIdx.x; i < LOC_; i += 256) mx = fmaxf(mx, p[i]);
  sd[threadIdx.x] = mx; __syncthreads();
  for (int s = 128; s > 0; s >>= 1) {
    if (threadIdx.x < s) sd[threadIdx.x] = fmaxf(sd[threadIdx.x], sd[threadIdx.x + s]);
    __syncthreads();
  }
  mx = sd[0]; __syncthreads();
  float sum = 0.f;
  for (int i = threadIdx.x; i < LOC_; i += 256) sum += __expf(p[i] - mx);
  sd[threadIdx.x] = sum; __syncthreads();
  for (int s = 128; s > 0; s >>= 1) {
    if (threadIdx.x < s) sd[threadIdx.x] += sd[threadIdx.x + s];
    __syncthreads();
  }
  if (threadIdx.x == 0) { stats[row * 2] = mx; stats[row * 2 + 1] = __logf(sd[0]); }
}

__global__ void k_finalize(float* __restrict__ Y, const float* __restrict__ stats) {
  const long long n = (long long)B_ * MAXT_ * LOC_;
  long long i  = (long long)blockIdx.x * blockDim.x + threadIdx.x;
  long long st = (long long)gridDim.x * blockDim.x;
  for (; i < n; i += st) {
    long long row = i / LOC_;
    Y[i] = Y[i] - stats[row * 2] - stats[row * 2 + 1];
  }
}

// ---------------------------------------------------------------------------
extern "C" void kernel_launch(void* const* d_in, const int* in_sizes, int n_in,
                              void* d_out, int out_size, void* d_ws, size_t ws_size,
                              hipStream_t stream) {
  (void)in_sizes; (void)n_in; (void)out_size; (void)ws_size;
  const int*   loc  = (const int*)d_in[0];
  const int*   tim  = (const int*)d_in[1];
  const int*   uid  = (const int*)d_in[2];
  const int*   tlen = (const int*)d_in[3];
  const int*   slen = (const int*)d_in[4];
  const float* eloc = (const float*)d_in[5];
  const float* etim = (const float*)d_in[6];
  const float* eusr = (const float*)d_in[7];
  const float* eWih = (const float*)d_in[8];
  const float* eWhh = (const float*)d_in[9];
  const float* ebih = (const float*)d_in[10];
  const float* ebhh = (const float*)d_in[11];
  const float* dWih = (const float*)d_in[12];
  const float* dWhh = (const float*)d_in[13];
  const float* dbih = (const float*)d_in[14];
  const float* dbhh = (const float*)d_in[15];
  const float* fcW  = (const float*)d_in[16];
  const float* fcb  = (const float*)d_in[17];

  float* score = (float*)d_out;                                    // 8192 x 20000
  float* omask = score + (long long)B_ * MAXT_ * LOC_;             // 8192

  // ---- workspace carving ----
  char* w = (char*)d_ws;
  size_t off = 0;
  auto carve = [&](size_t bytes) -> void* {
    void* p = w + off;
    off += (bytes + 255) & ~(size_t)255;
    return p;
  };
  bf16_t* xh     = (bf16_t*)carve((size_t)B_ * MAXH_ * D_ * 2);
  bf16_t* xc     = (bf16_t*)carve((size_t)B_ * MAXT_ * D_ * 2);
  bf16_t* WihE   = (bf16_t*)carve((size_t)G3_ * D_ * 2);
  bf16_t* WihD   = (bf16_t*)carve((size_t)G3_ * D_ * 2);
  bf16_t* WhhE   = (bf16_t*)carve((size_t)G3_ * H_ * 2);
  bf16_t* WhhD   = (bf16_t*)carve((size_t)G3_ * H_ * 2);
  bf16_t* fcWb   = (bf16_t*)carve((size_t)NPAD_ * 2 * H_ * 2);
  float*  gxE    = (float*)carve((size_t)B_ * MAXH_ * G3_ * 4);
  float*  gxD    = (float*)carve((size_t)B_ * MAXT_ * G3_ * 4);
  bf16_t* hhbf   = (bf16_t*)carve((size_t)B_ * MAXH_ * H_ * 2);
  bf16_t* hsbf   = (bf16_t*)carve((size_t)B_ * MAXT_ * H_ * 2);
  bf16_t* fcin   = (bf16_t*)carve((size_t)B_ * MAXT_ * 2 * H_ * 2);
  float*  stats  = (float*)carve((size_t)B_ * MAXT_ * 2 * 4);

  // ---- K0: weight conversions ----
  k_convert<<<1024, 256, 0, stream>>>(eWih, WihE, (long long)G3_ * D_);
  k_convert<<<1024, 256, 0, stream>>>(dWih, WihD, (long long)G3_ * D_);
  k_convert<<<1024, 256, 0, stream>>>(eWhh, WhhE, (long long)G3_ * H_);
  k_convert<<<1024, 256, 0, stream>>>(dWhh, WhhD, (long long)G3_ * H_);
  k_convert_pad<<<4096, 256, 0, stream>>>(fcW, fcWb, LOC_, NPAD_, 2 * H_);

  // ---- K1: embeddings + curr_mask ----
  k_embed<<<8192, 256, 0, stream>>>(loc, tim, uid, slen, tlen, eloc, etim, eusr, xh, xc);
  k_mask<<<(B_ * MAXT_ + 255) / 256, 256, 0, stream>>>(tlen, omask);

  // ---- K2: batched input GEMMs gx = x @ Wih^T (4 M-tiles per wave) ----
  k_gemm_xw<<<dim3(G3_ / 128, (B_ * MAXH_) / 64), 256, 0, stream>>>(xh, WihE, gxE);
  k_gemm_xw<<<dim3(G3_ / 128, (B_ * MAXT_) / 64), 256, 0, stream>>>(xc, WihD, gxD);

  // ---- K3: sequential GRUs (4 WGs x 16 batch rows; 147456 B dynamic LDS) ----
  size_t gru_lds = (size_t)16 * H_ * 4 + (size_t)16 * G3_ * 4 + (size_t)16 * H_ * 2;
  k_gru<<<B_ / 16, 256, gru_lds, stream>>>(gxE, WhhE, ebih, ebhh, slen, tlen,
                                           hhbf, MAXH_, 1);
  k_gru<<<B_ / 16, 256, gru_lds, stream>>>(gxD, WhhD, dbih, dbhh, slen, tlen,
                                           hsbf, MAXT_, 0);

  // ---- K4: attention + fcin assembly (196608 B dynamic LDS) ----
  size_t attn_lds = (size_t)MAXT_ * MAXH_ * 4 + (size_t)MAXT_ * MAXH_ * 2;
  k_attn<<<B_, 256, attn_lds, stream>>>(hhbf, hsbf, slen, tlen, fcin);

  // ---- K5: output projection (4 M-tiles per wave) ----
  k_fc<<<dim3(NPAD_ / 128, (B_ * MAXT_) / 64), 256, 0, stream>>>(fcin, fcWb, fcb, score);

  // ---- K6/K7: log-softmax ----
  k_lse<<<B_ * MAXT_, 256, 0, stream>>>(score, stats);
  k_finalize<<<8192, 256, 0, stream>>>(score, stats);
}
